// RetinaNet_38826504356571
// MI455X (gfx1250) — matrix operand order repaired and checked by reading, not compiled, not run on hardware
//
#include <hip/hip_runtime.h>

typedef __attribute__((ext_vector_type(16))) __bf16 bf16x16;
typedef __attribute__((ext_vector_type(2)))  __bf16 bf16x2;
typedef __attribute__((ext_vector_type(8)))  float  f32x8;

#define CIN    256
#define KTOT   (CIN * 9)     // 2304
#define KCH    72            // 2304 / 32 K-chunks
#define TILE_M 128
#define TILE_N 128
#define A_STRIDE 17          // padded row stride (uints) for A tile: 128 x 16 words
#define B_STRIDE 132         // padded row stride (uints) for B tile: 16 x 128 words

union FragBF { bf16x16 v; unsigned u[8]; };

__device__ __forceinline__ unsigned pack_bf16(float a, float b) {
#if __has_builtin(__builtin_amdgcn_cvt_pk_bf16_f32)
  auto t = __builtin_amdgcn_cvt_pk_bf16_f32(a, b);
  unsigned u;
  __builtin_memcpy(&u, &t, sizeof(u));
  return u;
#else
  union { bf16x2 v; unsigned u; } r;
  r.v[0] = (__bf16)a;
  r.v[1] = (__bf16)b;
  return r.u;
#endif
}

// Implicit-GEMM 3x3 SAME conv, K ordered tap-major: k = tap*CIN + cin.
// out[Cout, 8*H*W] = W[Cout, K] * im2col(in)[K, 8*H*W], bf16 WMMA, f32 accum.
// Block tile 128x128; 8 waves as 4(M) x 2(N); each wave: 2x4 = 8 WMMA per chunk.
// Double-buffered LDS software pipeline: 1 barrier per K-chunk, global loads
// for chunk kc+1 issued before the WMMAs of chunk kc.
// head_mode=0: out NCHW [8, Cout, H, W]
// head_mode=1: out[nb][cout/9][lvl_off + (cout%9)*HW + rem], group stride axis2
template <int LOG2W>
__global__ __launch_bounds__(256)
void retinanet_conv3x3_wmma(const float* __restrict__ in,
                            const float* __restrict__ wgt,   // [Cout, CIN, 3, 3]
                            const float* __restrict__ bias,  // [Cout]
                            float* __restrict__ out,
                            int Cout, int relu,
                            int head_mode, int axis2, int lvl_off) {
  constexpr int W      = 1 << LOG2W;
  constexpr int HW     = 1 << (2 * LOG2W);
  constexpr int totalP = 8 * HW;

  const int tid   = threadIdx.x;
  const int wave  = tid >> 5;
  const int lane  = tid & 31;
  const int mi    = wave >> 1;   // 0..3 : 32-row quarter of the 128-row tile
  const int ni    = wave & 1;    // 0..1 : 64-col half of the 128-col tile
  const int lhalf = lane >> 4;   // 0/1
  const int l16   = lane & 15;

  const int cout_base = blockIdx.x * TILE_M;
  const int p_base    = blockIdx.y * TILE_N;

  __shared__ unsigned a_lds[2][TILE_M * A_STRIDE];   // [m][k/2] packed bf16 pairs
  __shared__ unsigned b_lds[2][16 * B_STRIDE];       // [k/2][n] packed bf16 pairs

  // ---- per-thread fill coordinates (loop-invariant shifts/masks) ----
  const int  n_thr = tid & 127;
  const int  kk2b  = tid >> 7;                       // 0/1
  int pB = p_base + n_thr;
  const bool pOK = pB < totalP;                      // statically true (totalP % 128 == 0)
  if (pB > totalP - 1) pB = totalP - 1;
  const int  nbB = pB >> (2 * LOG2W);
  const int  remB = pB & (HW - 1);
  const int  ryB = remB >> LOG2W;
  const int  rxB = remB & (W - 1);
  const float* inN = in + (size_t)nbB * CIN * HW;
  const int  mA = tid >> 4;                          // A row (i adds 16*i)
  const int  kA = tid & 15;                          // A packed-K column

  // staged chunk values (registers)
  float av[8][2], bv[8][2];
  bool  okF = false;

  auto loadChunk = [&](int kc) {
    const int tap = kc >> 3;
    const int cb  = kc & 7;
    const int cin0 = cb * 32;
    const int t3 = tap / 3;
    const int dy = t3 - 1;
    const int dx = (tap - t3 * 3) - 1;
    const int y = ryB + dy;
    const int x = rxB + dx;
    okF = pOK && ((unsigned)y < (unsigned)W) && ((unsigned)x < (unsigned)W);
    int sp = y * W + x;                              // clamp: always-safe address
    sp = sp < 0 ? 0 : (sp > HW - 1 ? HW - 1 : sp);
    #pragma unroll
    for (int i = 0; i < 8; ++i) {                    // A: clamped row, no guard
      int cout = cout_base + mA + 16 * i;
      if (cout > Cout - 1) cout = Cout - 1;          // garbage rows masked at store
      const float* wp = wgt + (size_t)cout * KTOT + (size_t)(cin0 + 2 * kA) * 9 + tap;
      av[i][0] = wp[0];
      av[i][1] = wp[9];                              // next cin, same tap
    }
    #pragma unroll
    for (int i = 0; i < 8; ++i) {                    // B: always load, select later
      int cin = cin0 + 2 * (kk2b + 2 * i);
      const float* sp2 = inN + (size_t)cin * HW + sp;
      bv[i][0] = sp2[0];
      bv[i][1] = sp2[HW];                            // next cin, channel stride
    }
  };

  auto storeChunk = [&](int buf) {
    #pragma unroll
    for (int i = 0; i < 8; ++i)
      a_lds[buf][(mA + 16 * i) * A_STRIDE + kA] = pack_bf16(av[i][0], av[i][1]);
    #pragma unroll
    for (int i = 0; i < 8; ++i) {
      unsigned w = pack_bf16(bv[i][0], bv[i][1]);
      b_lds[buf][(kk2b + 2 * i) * B_STRIDE + n_thr] = okF ? w : 0u;  // 1 cndmask
    }
  };

  f32x8 acc[2][4] = {};

  loadChunk(0);
  storeChunk(0);

  #pragma unroll 1
  for (int kc = 0; kc < KCH; ++kc) {
    __syncthreads();                                 // single barrier per chunk
    const int buf = kc & 1;
    const bool more = (kc + 1) < KCH;
    if (more) loadChunk(kc + 1);                     // global loads fly under WMMAs

    FragBF A0, A1, B[4];
    #pragma unroll
    for (int j = 0; j < 8; ++j) {
      int kp = ((j < 4) ? 0 : 8) + lhalf * 4 + (j & 3);
      A0.u[j] = a_lds[buf][(mi * 32 + l16) * A_STRIDE + kp];
      A1.u[j] = a_lds[buf][(mi * 32 + 16 + l16) * A_STRIDE + kp];
    }
    #pragma unroll
    for (int nn = 0; nn < 4; ++nn) {
      #pragma unroll
      for (int j = 0; j < 8; ++j) {
        int kp = ((j < 4) ? 0 : 8) + lhalf * 4 + (j & 3);
        B[nn].u[j] = b_lds[buf][kp * B_STRIDE + ni * 64 + nn * 16 + l16];
      }
    }
    #pragma unroll
    for (int nn = 0; nn < 4; ++nn) {
      acc[0][nn] = __builtin_amdgcn_wmma_f32_16x16x32_bf16(false, A0.v, false, B[nn].v,
                                                           (short)0, acc[0][nn], false, false);
      acc[1][nn] = __builtin_amdgcn_wmma_f32_16x16x32_bf16(false, A1.v, false, B[nn].v,
                                                           (short)0, acc[1][nn], false, false);
    }

    if (more) storeChunk(buf ^ 1);                   // stage next chunk
  }

  // ---- epilogue: bias (+ReLU), store. D layout: lane = N, VGPR r = M ----
  const int G = head_mode ? (Cout / 9) : 0;
  #pragma unroll
  for (int nn = 0; nn < 4; ++nn) {
    int p = p_base + ni * 64 + nn * 16 + l16;
    if (p < totalP) {
      int nb  = p >> (2 * LOG2W);
      int rem = p & (HW - 1);
      #pragma unroll
      for (int mm = 0; mm < 2; ++mm) {
        #pragma unroll
        for (int r = 0; r < 8; ++r) {
          int cout = cout_base + mi * 32 + mm * 16 + lhalf * 8 + r;
          if (cout < Cout) {
            float v = acc[mm][nn][r] + bias[cout];
            if (relu) v = fmaxf(v, 0.f);
            size_t idx;
            if (head_mode) {
              int g  = cout / 9;
              int a9 = cout - g * 9;
              idx = ((size_t)nb * G + g) * (size_t)axis2 + (size_t)lvl_off
                  + (size_t)a9 * HW + rem;
            } else {
              idx = ((size_t)nb * Cout + cout) * (size_t)HW + rem;
            }
            out[idx] = v;
          }
        }
      }
    }
  }
}

static void launch_conv(int W, dim3 grid, hipStream_t stream,
                        const float* in, const float* wgt, const float* bias,
                        float* out, int Cout, int relu,
                        int head_mode, int axis2, int lvl_off) {
  dim3 block(256);
  switch (W) {
    case 64: retinanet_conv3x3_wmma<6><<<grid, block, 0, stream>>>(in, wgt, bias, out, Cout, relu, head_mode, axis2, lvl_off); break;
    case 32: retinanet_conv3x3_wmma<5><<<grid, block, 0, stream>>>(in, wgt, bias, out, Cout, relu, head_mode, axis2, lvl_off); break;
    case 16: retinanet_conv3x3_wmma<4><<<grid, block, 0, stream>>>(in, wgt, bias, out, Cout, relu, head_mode, axis2, lvl_off); break;
    case 8:  retinanet_conv3x3_wmma<3><<<grid, block, 0, stream>>>(in, wgt, bias, out, Cout, relu, head_mode, axis2, lvl_off); break;
    case 4:  retinanet_conv3x3_wmma<2><<<grid, block, 0, stream>>>(in, wgt, bias, out, Cout, relu, head_mode, axis2, lvl_off); break;
  }
}

extern "C" void kernel_launch(void* const* d_in, const int* in_sizes, int n_in,
                              void* d_out, int out_size, void* d_ws, size_t ws_size,
                              hipStream_t stream) {
  const float* feats[5];
  for (int i = 0; i < 5; ++i) feats[i] = (const float*)d_in[i];
  const float* reg_stem_w = (const float*)d_in[5];
  const float* reg_stem_b = (const float*)d_in[6];
  const float* reg_head_w = (const float*)d_in[7];
  const float* reg_head_b = (const float*)d_in[8];
  const float* cls_stem_w = (const float*)d_in[9];
  const float* cls_stem_b = (const float*)d_in[10];
  const float* cls_head_w = (const float*)d_in[11];
  const float* cls_head_b = (const float*)d_in[12];
  float* out = (float*)d_out;

  const int Hs[5] = {64, 32, 16, 8, 4};
  const int axis2 = 9 * (64 * 64 + 32 * 32 + 16 * 16 + 8 * 8 + 4 * 4);  // 49104
  const size_t conf_base = (size_t)8 * 4 * axis2;   // bbox block precedes conf block
  const size_t bufElems  = (size_t)8 * 256 * 64 * 64;
  float* wsA = (float*)d_ws;
  float* wsB = wsA + bufElems;

  int lvl_off = 0;
  for (int l = 0; l < 5; ++l) {
    const int W = Hs[l], HW = W * W;
    const int totalP = 8 * HW;
    const int spT = (totalP + TILE_N - 1) / TILE_N;

    for (int s = 0; s < 2; ++s) {  // 0 = reg subnet, 1 = cls subnet
      const float* sw = s ? cls_stem_w : reg_stem_w;
      const float* sb = s ? cls_stem_b : reg_stem_b;
      const float* hw = s ? cls_head_w : reg_head_w;
      const float* hb = s ? cls_head_b : reg_head_b;
      const int CoutH = s ? 81 : 36;

      const float* cur = feats[l];
      float* dst = wsA;
      for (int i = 0; i < 4; ++i) {
        dim3 grid(256 / TILE_M, spT);
        launch_conv(W, grid, stream, cur,
                    sw + (size_t)i * 256 * KTOT, sb + (size_t)i * 256,
                    dst, 256, /*relu=*/1, /*head=*/0, 0, 0);
        cur = dst;
        dst = (dst == wsA) ? wsB : wsA;
      }

      const int mT = (CoutH + TILE_M - 1) / TILE_M;
      float* headOut = s ? (out + conf_base) : out;
      dim3 grid(mT, spT);
      launch_conv(W, grid, stream, cur, hw, hb, headOut, CoutH,
                  /*relu=*/0, /*head=*/1, axis2, lvl_off);
    }
    lvl_off += 9 * HW;
  }
}
